// SowinTransformerBlock_21852793602096
// MI455X (gfx1250) — matrix operand order repaired and checked by reading, not compiled, not run on hardware
//
#include <hip/hip_runtime.h>
#include <hip/hip_bf16.h>

typedef _Float16 v16h __attribute__((ext_vector_type(16)));
typedef _Float16 v8h  __attribute__((ext_vector_type(8)));
typedef float    v8f  __attribute__((ext_vector_type(8)));
typedef int      v4i  __attribute__((ext_vector_type(4)));

#define SHIFT_ 4

#if defined(__has_builtin)
#if __has_builtin(__builtin_amdgcn_global_load_async_to_lds_b128)
#define HAVE_ASYNC_LDS 1
#endif
#endif
#ifndef HAVE_ASYNC_LDS
#define HAVE_ASYNC_LDS 0
#endif

#if HAVE_ASYNC_LDS
// The builtin's parameters are v4i* in AS1 (global) and AS3 (LDS):
typedef __attribute__((address_space(1))) v4i gbl_v4i;
typedef __attribute__((address_space(3))) v4i lds_v4i;
#endif

// ---------------------------------------------------------------------------
// Fragment loaders for V_WMMA_F32_16X16X32_F16 (wave32).
// A (16x32, MxK) row-major, ld in halves:
//   lane: m = lane&15, g = lane>>4; halves [0..7] = K[g*8 .. g*8+7],
//   halves [8..15] = K[16+g*8 .. 16+g*8+7].   (ISA 7.12.2 16-bit A table)
// B (32x16, KxN) supplied as B^T row-major (NxK, ld in halves):
//   lane: n = lane&15, g = lane>>4; halves [0..7] = K[g*8 .. g*8+7],
//   halves [8..15] = K[16+g*8 .. 16+g*8+7].
// ---------------------------------------------------------------------------
__device__ __forceinline__ v16h load_fragA(const _Float16* base, int row0, int k0, int ld) {
  int lane = threadIdx.x & 31;
  int m = lane & 15, g = lane >> 4;
  const _Float16* p = base + (size_t)(row0 + m) * ld + k0 + g * 8;
  v8h lo = *reinterpret_cast<const v8h*>(p);
  v8h hi = *reinterpret_cast<const v8h*>(p + 16);
  v16h o;
#pragma unroll
  for (int i = 0; i < 8; ++i) { o[i] = lo[i]; o[8 + i] = hi[i]; }
  return o;
}

__device__ __forceinline__ v16h load_fragB(const _Float16* baseT, int col0, int k0, int ld) {
  int lane = threadIdx.x & 31;
  int n = lane & 15, g = lane >> 4;
  const _Float16* p = baseT + (size_t)(col0 + n) * ld + k0 + g * 8;
  v8h lo = *reinterpret_cast<const v8h*>(p);
  v8h hi = *reinterpret_cast<const v8h*>(p + 16);
  v16h o;
#pragma unroll
  for (int i = 0; i < 8; ++i) { o[i] = lo[i]; o[8 + i] = hi[i]; }
  return o;
}

#define WMMA_F16(a, b, c) \
  __builtin_amdgcn_wmma_f32_16x16x32_f16(false, (a), false, (b), (short)0, (c), false, false)

// ---------------------------------------------------------------------------
// Weight fp32 -> fp16 conversion (weights then live in L2 for all blocks)
// ---------------------------------------------------------------------------
__global__ void k_cvt(const float* __restrict__ src, _Float16* __restrict__ dst, int n) {
  int i = blockIdx.x * 256 + threadIdx.x;
  if (i < n) dst[i] = (_Float16)src[i];
}

// ---------------------------------------------------------------------------
// K1: LN1 + cyclic shift + window partition + QKV GEMM (64x576x192)
// One block per window (2048 blocks, 256 threads = 8 waves)
// ---------------------------------------------------------------------------
__global__ __launch_bounds__(256) void k_ln1_qkv(
    const float* __restrict__ x, const float* __restrict__ g1,
    const float* __restrict__ b1, const float* __restrict__ qkv_b,
    const _Float16* __restrict__ qkvw16,
    _Float16* __restrict__ Qh, _Float16* __restrict__ Kh, _Float16* __restrict__ Vt) {
  __shared__ __align__(16) float xf[64 * 192];
  __shared__ __align__(16) _Float16 xh[64 * 192];
  __shared__ float ssum[64], ssq[64];
  int tid = threadIdx.x;
  int w = blockIdx.x;
  int b = w >> 10, wi = w & 1023;
  int wh = wi >> 5, wc = wi & 31;

  if (tid < 64) { ssum[tid] = 0.f; ssq[tid] = 0.f; }
  __syncthreads();
  for (int e = tid; e < 64 * 192; e += 256) {
    int t = e / 192, c = e - t * 192;
    int tr = t >> 3, tc = t & 7;
    int i = (wh * 8 + tr + SHIFT_) & 255;   // roll(-SHIFT) fused into gather
    int j = (wc * 8 + tc + SHIFT_) & 255;
    float v = x[((size_t)b * 65536 + i * 256 + j) * 192 + c];
    xf[e] = v;
    atomicAdd(&ssum[t], v);
    atomicAdd(&ssq[t], v * v);
  }
  __syncthreads();
  for (int e = tid; e < 64 * 192; e += 256) {
    int t = e / 192, c = e - t * 192;
    float m = ssum[t] * (1.f / 192.f);
    float var = ssq[t] * (1.f / 192.f) - m * m;
    float rs = rsqrtf(var + 1e-5f);
    xh[e] = (_Float16)((xf[e] - m) * rs * g1[c] + b1[c]);
  }
  __syncthreads();

  int wv = tid >> 5, lane = tid & 31;
  int nl = lane & 15, g = lane >> 4;
  for (int tile = wv; tile < 144; tile += 8) {     // 4 m-tiles x 36 n-tiles
    int mt = tile & 3, nt = tile >> 2;
    v8f acc = {};
#pragma unroll
    for (int k0 = 0; k0 < 192; k0 += 32) {
      v16h a = load_fragA(xh, mt * 16, k0, 192);
      v16h bb = load_fragB(qkvw16, nt * 16, k0, 192);  // qkv_w rows are B^T rows
      acc = WMMA_F16(a, bb, acc);
    }
    // q/k/v segment + head are wave-uniform per tile: pin them to SGPRs so the
    // three-way store select is a scalar branch, not exec-mask divergence.
    int s    = __builtin_amdgcn_readfirstlane(nt / 12);        // 192/16=12 tiles/seg
    int ntl  = nt - s * 12;
    int head = __builtin_amdgcn_readfirstlane(ntl >> 1);
    int dh   = ((ntl & 1) << 4) + nl;
    int o    = nt * 16 + nl;                                   // bias column
    size_t hb = (size_t)(w * 6 + head);
#pragma unroll
    for (int r = 0; r < 8; ++r) {
      int m = mt * 16 + r + 8 * g;
      float val = acc[r] + qkv_b[o];
      if (s == 0)      Qh[(hb * 64 + m) * 32 + dh] = (_Float16)(val * 0.17677669529663689f);
      else if (s == 1) Kh[(hb * 64 + m) * 32 + dh] = (_Float16)val;
      else             Vt[(hb * 32 + dh) * 64 + m] = (_Float16)val;   // V transposed
    }
  }
}

// ---------------------------------------------------------------------------
// K2: windowed attention. One (window, head) per block (12288 blocks,
// 128 threads = 4 waves, each owns a 16-row strip).
// ---------------------------------------------------------------------------
__global__ __launch_bounds__(128) void k_attn(
    const _Float16* __restrict__ Qh, const _Float16* __restrict__ Kh,
    const _Float16* __restrict__ Vt, const float* __restrict__ rpb,
    const int* __restrict__ rel_idx, const float* __restrict__ mask,
    _Float16* __restrict__ attn_out) {
  __shared__ __align__(16) _Float16 plds[4 * 16 * 64];
  int tid = threadIdx.x, lane = tid & 31, wv = tid >> 5;
  int nl = lane & 15, g = lane >> 4;
  int bid = blockIdx.x;
  int w = bid / 6, head = bid - w * 6;
  int wi = w & 1023;
  const _Float16* q  = Qh + (size_t)bid * 64 * 32;
  const _Float16* kk = Kh + (size_t)bid * 64 * 32;
  const _Float16* vt = Vt + (size_t)bid * 32 * 64;
  int m0 = wv * 16;

  // S = (Q*scale) K^T : K-dim = DH = 32 -> single WMMA per 16x16 tile
  v16h a = load_fragA(q, m0, 0, 32);
  v8f sc[4];
#pragma unroll
  for (int ct = 0; ct < 4; ++ct) {
    v16h bb = load_fragB(kk, ct * 16, 0, 32);       // K rows == (K^T)^T rows
    v8f z = {};
    sc[ct] = WMMA_F16(a, bb, z);
  }
  const float* mrow = mask + (size_t)wi * 4096;
#pragma unroll
  for (int ct = 0; ct < 4; ++ct) {
    int n = ct * 16 + nl;
#pragma unroll
    for (int r = 0; r < 8; ++r) {
      int m = m0 + r + 8 * g;
      sc[ct][r] += rpb[rel_idx[m * 64 + n] * 6 + head] + mrow[m * 64 + n];
    }
  }
  // softmax over each row (64 values spread over 4 tiles x 16 lanes)
  float inv[8];
#pragma unroll
  for (int r = 0; r < 8; ++r) {
    float v = fmaxf(fmaxf(sc[0][r], sc[1][r]), fmaxf(sc[2][r], sc[3][r]));
    v = fmaxf(v, __shfl_xor(v, 1));
    v = fmaxf(v, __shfl_xor(v, 2));
    v = fmaxf(v, __shfl_xor(v, 4));
    v = fmaxf(v, __shfl_xor(v, 8));
    float s = 0.f;
#pragma unroll
    for (int ct = 0; ct < 4; ++ct) { sc[ct][r] = __expf(sc[ct][r] - v); s += sc[ct][r]; }
    s += __shfl_xor(s, 1); s += __shfl_xor(s, 2);
    s += __shfl_xor(s, 4); s += __shfl_xor(s, 8);
    inv[r] = 1.f / s;
  }
  _Float16* pw = plds + wv * 16 * 64;
#pragma unroll
  for (int ct = 0; ct < 4; ++ct) {
    int n = ct * 16 + nl;
#pragma unroll
    for (int r = 0; r < 8; ++r)
      pw[(r + 8 * g) * 64 + n] = (_Float16)(sc[ct][r] * inv[r]);
  }
  __syncthreads();

  // O = P V  (K-dim = 64 -> 2 k-steps; B^T rows are Vt rows, contiguous)
#pragma unroll
  for (int nt = 0; nt < 2; ++nt) {
    v8f acc = {};
#pragma unroll
    for (int k0 = 0; k0 < 64; k0 += 32) {
      v16h pa = load_fragA(pw, 0, k0, 64);
      v16h vb = load_fragB(vt, nt * 16, k0, 64);
      acc = WMMA_F16(pa, vb, acc);
    }
#pragma unroll
    for (int r = 0; r < 8; ++r) {
      int m = m0 + r + 8 * g;
      int dh = nt * 16 + nl;
      attn_out[((size_t)w * 64 + m) * 192 + head * 32 + dh] = (_Float16)acc[r];
    }
  }
}

// ---------------------------------------------------------------------------
// K3: proj GEMM (64x192x192) + window reverse + roll(+SHIFT) + residual
// ---------------------------------------------------------------------------
__global__ __launch_bounds__(256) void k_proj(
    const _Float16* __restrict__ ao, const _Float16* __restrict__ projw16,
    const float* __restrict__ proj_b, const float* __restrict__ x,
    float* __restrict__ x1) {
  int tid = threadIdx.x, lane = tid & 31, wv = tid >> 5;
  int nl = lane & 15, g = lane >> 4;
  int w = blockIdx.x;
  int b = w >> 10, wi = w & 1023;
  int wh = wi >> 5, wc = wi & 31;
  const _Float16* A = ao + (size_t)w * 64 * 192;
  for (int tile = wv; tile < 48; tile += 8) {
    int mt = tile & 3, nt = tile >> 2;
    v8f acc = {};
#pragma unroll
    for (int k0 = 0; k0 < 192; k0 += 32) {
      v16h a = load_fragA(A, mt * 16, k0, 192);
      v16h bb = load_fragB(projw16, nt * 16, k0, 192);
      acc = WMMA_F16(a, bb, acc);
    }
    int c = nt * 16 + nl;
#pragma unroll
    for (int r = 0; r < 8; ++r) {
      int m = mt * 16 + r + 8 * g;
      int tr = m >> 3, tc = m & 7;
      int i = (wh * 8 + tr + SHIFT_) & 255;   // roll(+SHIFT) scatter
      int j = (wc * 8 + tc + SHIFT_) & 255;
      size_t tok = (size_t)b * 65536 + i * 256 + j;
      x1[tok * 192 + c] = x[tok * 192 + c] + acc[r] + proj_b[c];
    }
  }
}

// ---------------------------------------------------------------------------
// K4: LN2 + FC1 GEMM (64x768x192) + exact GELU -> h2 (f16)
// ---------------------------------------------------------------------------
__global__ __launch_bounds__(256) void k_ln2_fc1(
    const float* __restrict__ x1, const float* __restrict__ g2,
    const float* __restrict__ b2, const _Float16* __restrict__ fc1w16,
    const float* __restrict__ fc1_b, _Float16* __restrict__ h2) {
  __shared__ __align__(16) float xf[64 * 192];
  __shared__ __align__(16) _Float16 xh[64 * 192];
  __shared__ float ssum[64], ssq[64];
  int tid = threadIdx.x;
  size_t tok0 = (size_t)blockIdx.x * 64;
  if (tid < 64) { ssum[tid] = 0.f; ssq[tid] = 0.f; }
  __syncthreads();
  for (int e = tid; e < 64 * 192; e += 256) {
    int t = e / 192;
    float v = x1[tok0 * 192 + e];
    xf[e] = v;
    atomicAdd(&ssum[t], v);
    atomicAdd(&ssq[t], v * v);
  }
  __syncthreads();
  for (int e = tid; e < 64 * 192; e += 256) {
    int t = e / 192, c = e - t * 192;
    float m = ssum[t] * (1.f / 192.f);
    float var = ssq[t] * (1.f / 192.f) - m * m;
    xh[e] = (_Float16)((xf[e] - m) * rsqrtf(var + 1e-5f) * g2[c] + b2[c]);
  }
  __syncthreads();

  int wv = tid >> 5, lane = tid & 31, nl = lane & 15, g = lane >> 4;
  for (int tile = wv; tile < 192; tile += 8) {     // 4 x 48 tiles
    int mt = tile & 3, nt = tile >> 2;
    v8f acc = {};
#pragma unroll
    for (int k0 = 0; k0 < 192; k0 += 32) {
      v16h a = load_fragA(xh, mt * 16, k0, 192);
      v16h bb = load_fragB(fc1w16, nt * 16, k0, 192);
      acc = WMMA_F16(a, bb, acc);
    }
    int n = nt * 16 + nl;
#pragma unroll
    for (int r = 0; r < 8; ++r) {
      int m = mt * 16 + r + 8 * g;
      float u = acc[r] + fc1_b[n];
      float gl = 0.5f * u * (1.f + erff(u * 0.70710678118654752f));  // exact GELU
      h2[(tok0 + m) * 768 + n] = (_Float16)gl;
    }
  }
}

// ---------------------------------------------------------------------------
// K5: FC2 GEMM (64x192x768) + residual -> output (f32)
// A-block (64x768 f16 = 96KB) staged to LDS via async global->LDS DMA so each
// row is read from L2 once instead of 12x; drained with s_wait_asynccnt.
// ---------------------------------------------------------------------------
__global__ __launch_bounds__(256) void k_fc2(
    const _Float16* __restrict__ h2, const _Float16* __restrict__ fc2w16,
    const float* __restrict__ fc2_b, const float* __restrict__ x1,
    float* __restrict__ out) {
  __shared__ __align__(16) _Float16 ah[64 * 768];
  int tid = threadIdx.x, lane = tid & 31, wv = tid >> 5;
  int nl = lane & 15, g = lane >> 4;
  size_t tok0 = (size_t)blockIdx.x * 64;
  const _Float16* A = h2 + tok0 * 768;

#if HAVE_ASYNC_LDS
#pragma unroll 4
  for (int e = tid * 8; e < 64 * 768; e += 256 * 8) {
    __builtin_amdgcn_global_load_async_to_lds_b128(
        (gbl_v4i*)(A + e), (lds_v4i*)(ah + e), 0, 0);
  }
  asm volatile("s_wait_asynccnt 0" ::: "memory");   // drain ASYNCcnt (ISA 5.7.2)
#else
  for (int e = tid * 8; e < 64 * 768; e += 256 * 8)
    *reinterpret_cast<v8h*>(ah + e) = *reinterpret_cast<const v8h*>(A + e);
#endif
  __syncthreads();

  for (int tile = wv; tile < 48; tile += 8) {
    int mt = tile & 3, nt = tile >> 2;
    v8f acc = {};
#pragma unroll
    for (int k0 = 0; k0 < 768; k0 += 32) {
      v16h a = load_fragA(ah, mt * 16, k0, 768);
      v16h bb = load_fragB(fc2w16, nt * 16, k0, 768);
      acc = WMMA_F16(a, bb, acc);
    }
    int c = nt * 16 + nl;
#pragma unroll
    for (int r = 0; r < 8; ++r) {
      int m = mt * 16 + r + 8 * g;
      size_t tok = tok0 + m;
      out[tok * 192 + c] = x1[tok * 192 + c] + acc[r] + fc2_b[c];
    }
  }
}

// ---------------------------------------------------------------------------
// Launcher
// ---------------------------------------------------------------------------
extern "C" void kernel_launch(void* const* d_in, const int* in_sizes, int n_in,
                              void* d_out, int out_size, void* d_ws, size_t ws_size,
                              hipStream_t stream) {
  (void)in_sizes; (void)n_in; (void)out_size; (void)ws_size;
  const float* x      = (const float*)d_in[0];
  const float* ln1_g  = (const float*)d_in[1];
  const float* ln1_b  = (const float*)d_in[2];
  const float* qkv_w  = (const float*)d_in[3];
  const float* qkv_b  = (const float*)d_in[4];
  const float* rpb    = (const float*)d_in[5];
  const float* proj_w = (const float*)d_in[6];
  const float* proj_b = (const float*)d_in[7];
  const float* ln2_g  = (const float*)d_in[8];
  const float* ln2_b  = (const float*)d_in[9];
  const float* fc1_w  = (const float*)d_in[10];
  const float* fc1_b  = (const float*)d_in[11];
  const float* fc2_w  = (const float*)d_in[12];
  const float* fc2_b  = (const float*)d_in[13];
  const float* amask  = (const float*)d_in[14];
  const int*   rel    = (const int*)d_in[15];
  float* out = (float*)d_out;
  char* ws = (char*)d_ws;

  // ws layout (all offsets 256B-aligned)
  _Float16* qkvw16  = (_Float16*)(ws + 0);          // 576*192   f16
  _Float16* projw16 = (_Float16*)(ws + 221184);     // 192*192   f16
  _Float16* fc1w16  = (_Float16*)(ws + 294912);     // 768*192   f16
  _Float16* fc2w16  = (_Float16*)(ws + 589824);     // 192*768   f16
  _Float16* Qh      = (_Float16*)(ws + 884736);     // 2048*6*64*32 f16
  _Float16* Kh      = (_Float16*)(ws + 51216384);
  _Float16* Vt      = (_Float16*)(ws + 101548032);  // transposed V
  _Float16* ao      = (_Float16*)(ws + 151879680);  // 2048*64*192 f16
  float*    x1      = (float*)(ws + 202211328);     // 131072*192 f32
  _Float16* h2      = (_Float16*)(ws + 302874624);  // 131072*768 f16

  k_cvt<<<(110592 + 255) / 256, 256, 0, stream>>>(qkv_w, qkvw16, 110592);
  k_cvt<<<(36864 + 255) / 256, 256, 0, stream>>>(proj_w, projw16, 36864);
  k_cvt<<<(147456 + 255) / 256, 256, 0, stream>>>(fc1_w, fc1w16, 147456);
  k_cvt<<<(147456 + 255) / 256, 256, 0, stream>>>(fc2_w, fc2w16, 147456);

  k_ln1_qkv<<<2048, 256, 0, stream>>>(x, ln1_g, ln1_b, qkv_b, qkvw16, Qh, Kh, Vt);
  k_attn<<<12288, 128, 0, stream>>>(Qh, Kh, Vt, rpb, rel, amask, ao);
  k_proj<<<2048, 256, 0, stream>>>(ao, projw16, proj_b, x, x1);
  k_ln2_fc1<<<2048, 256, 0, stream>>>(x1, ln2_g, ln2_b, fc1w16, fc1_b, h2);
  k_fc2<<<2048, 256, 0, stream>>>(h2, fc2w16, fc2_b, x1, out);
}
